// PSM_34342558499368
// MI455X (gfx1250) — matrix operand order repaired and checked by reading, not compiled, not run on hardware
//
#include <hip/hip_runtime.h>
#include <hip/hip_bf16.h>
#include <math.h>

// ---------------- CDNA5 WMMA types ----------------
typedef __attribute__((ext_vector_type(16))) __bf16 v16bf;
typedef __attribute__((ext_vector_type(8)))  float  v8f;

#define N_ROWS        262144
#define DDIM          64
#define KCODES        1024
#define ROWS_PER_WAVE 32                                 // two 16-row M-groups share each B fetch
#define WAVES_PER_BLK 4
#define ROWS_PER_BLK  (ROWS_PER_WAVE * WAVES_PER_BLK)    // 128
#define NBLOCKS       (N_ROWS / ROWS_PER_BLK)            // 2048
#define KTILES        (KCODES / 16)                      // 64

// RNE float -> bf16 bits (data is finite random normal; no NaN path needed)
static __device__ __forceinline__ unsigned short f2bf(float f) {
    unsigned u = __float_as_uint(f);
    return (unsigned short)((u + 0x7FFFu + ((u >> 16) & 1u)) >> 16);
}
static __device__ __forceinline__ float bf2f(unsigned short h) {
    return __uint_as_float(((unsigned)h) << 16);
}

// ------------- Phase 0: normalize codebook, build bf16 hi/lo split -------------
// grid = KCODES blocks of 32 (one wave per code row)
__global__ void vq_prep(const float* __restrict__ w,
                        float* __restrict__ wn, float* __restrict__ wn2,
                        unsigned short* __restrict__ whi, unsigned short* __restrict__ wlo) {
    const int k    = blockIdx.x;
    const int lane = threadIdx.x;            // 0..31
    float a = w[(size_t)k * DDIM + lane];
    float b = w[(size_t)k * DDIM + 32 + lane];
    float s = a * a + b * b;
    #pragma unroll
    for (int m = 1; m < 32; m <<= 1) s += __shfl_xor(s, m, 32);
    const float denom = fmaxf(sqrtf(s), 1e-12f);     // F.normalize: v / max(||v||, eps)
    const float na = a / denom, nb = b / denom;
    float s2 = na * na + nb * nb;                    // faithful ||wn||^2 (post-normalize)
    #pragma unroll
    for (int m = 1; m < 32; m <<= 1) s2 += __shfl_xor(s2, m, 32);

    wn[(size_t)k * DDIM + lane]      = na;
    wn[(size_t)k * DDIM + 32 + lane] = nb;
    const unsigned short ha = f2bf(na), hb = f2bf(nb);
    whi[(size_t)k * DDIM + lane]      = ha;
    whi[(size_t)k * DDIM + 32 + lane] = hb;
    wlo[(size_t)k * DDIM + lane]      = f2bf(na - bf2f(ha));
    wlo[(size_t)k * DDIM + 32 + lane] = f2bf(nb - bf2f(hb));
    if (lane == 0) wn2[k] = s2;
}

// ------------- Phase 1: normalize x, WMMA distance search, argmin, outputs -------------
__global__ __launch_bounds__(128) void vq_main(
        const float* __restrict__ x,
        const float* __restrict__ wn, const float* __restrict__ wn2,
        const unsigned short* __restrict__ whi, const unsigned short* __restrict__ wlo,
        float* __restrict__ outQ, float* __restrict__ outIdx,
        float* __restrict__ partial) {
    __shared__ float          lds_xn [ROWS_PER_BLK * DDIM];   // 32 KB
    __shared__ unsigned short lds_hi [ROWS_PER_BLK * DDIM];   // 16 KB
    __shared__ unsigned short lds_lo [ROWS_PER_BLK * DDIM];   // 16 KB
    __shared__ float          lds_xn2[ROWS_PER_BLK];
    __shared__ float          lds_ws [WAVES_PER_BLK];

    const int tid      = threadIdx.x;
    const int lane     = tid & 31;
    const int wave     = tid >> 5;
    const int lhalf    = lane >> 4;          // 0 | 1
    const int l16      = lane & 15;
    const int rowBlock = blockIdx.x * ROWS_PER_BLK;
    const int wr       = wave * ROWS_PER_WAVE;   // this wave's local row base (32 rows)

    // ---- normalize 32 rows of x per wave; 2 rows per iteration (16 lanes x float4 each) ----
    for (int i = 0; i < 16; ++i) {
        const int rl = wr + 2 * i + lhalf;
        const int rg = rowBlock + rl;
        const float4 xv = *(const float4*)(x + (size_t)rg * DDIM + l16 * 4);
        float s = xv.x * xv.x + xv.y * xv.y + xv.z * xv.z + xv.w * xv.w;
        s += __shfl_xor(s, 1, 32); s += __shfl_xor(s, 2, 32);
        s += __shfl_xor(s, 4, 32); s += __shfl_xor(s, 8, 32);   // stays within 16-lane half
        const float denom = fmaxf(sqrtf(s), 1e-12f);
        float4 nv;
        nv.x = xv.x / denom; nv.y = xv.y / denom; nv.z = xv.z / denom; nv.w = xv.w / denom;
        float s2 = nv.x * nv.x + nv.y * nv.y + nv.z * nv.z + nv.w * nv.w;
        s2 += __shfl_xor(s2, 1, 32); s2 += __shfl_xor(s2, 2, 32);
        s2 += __shfl_xor(s2, 4, 32); s2 += __shfl_xor(s2, 8, 32);
        const int off = rl * DDIM + l16 * 4;
        *(float4*)(lds_xn + off) = nv;
        const unsigned short h0 = f2bf(nv.x), h1 = f2bf(nv.y), h2 = f2bf(nv.z), h3 = f2bf(nv.w);
        lds_hi[off + 0] = h0; lds_hi[off + 1] = h1; lds_hi[off + 2] = h2; lds_hi[off + 3] = h3;
        lds_lo[off + 0] = f2bf(nv.x - bf2f(h0));
        lds_lo[off + 1] = f2bf(nv.y - bf2f(h1));
        lds_lo[off + 2] = f2bf(nv.z - bf2f(h2));
        lds_lo[off + 3] = f2bf(nv.w - bf2f(h3));
        if (l16 == 0) lds_xn2[rl] = s2;
    }
    __syncthreads();

    // ---- A matrices (16x32 bf16 layout, ISA 7.12.2): {hi,lo} x K-chunk x M-group ----
    union AV { v16bf v; uint4 q[2]; };
    AV aHi0[2], aHi1[2], aLo0[2], aLo1[2];
    #pragma unroll
    for (int g = 0; g < 2; ++g) {
        const int ab = (wr + g * 16 + l16) * DDIM + lhalf * 8;   // lane holds row M=lane%16
        aHi0[g].q[0] = *(const uint4*)(lds_hi + ab);      aHi0[g].q[1] = *(const uint4*)(lds_hi + ab + 16);
        aHi1[g].q[0] = *(const uint4*)(lds_hi + ab + 32); aHi1[g].q[1] = *(const uint4*)(lds_hi + ab + 48);
        aLo0[g].q[0] = *(const uint4*)(lds_lo + ab);      aLo0[g].q[1] = *(const uint4*)(lds_lo + ab + 16);
        aLo1[g].q[0] = *(const uint4*)(lds_lo + ab + 32); aLo1[g].q[1] = *(const uint4*)(lds_lo + ab + 48);
    }

    // per-slot row ||xn||^2 (C layout: vgpr j of group g -> row g*16 + j + 8*(lane>=16))
    float rx2[16];
    #pragma unroll
    for (int g = 0; g < 2; ++g)
        #pragma unroll
        for (int j = 0; j < 8; ++j)
            rx2[g * 8 + j] = lds_xn2[wr + g * 16 + lhalf * 8 + j];

    float bestD[16]; int bestI[16];
    #pragma unroll
    for (int j = 0; j < 16; ++j) { bestD[j] = 3.4028235e38f; bestI[j] = 0; }

    // ---- K loop: 64 code tiles of 16; 12 bf16 WMMAs per tile (bf16x3 split, 2 M-groups) ----
    const int bd0 = lhalf * 8;      // B layout: lane holds col N=lane%16, K runs 8h..8h+7 and 16+8h..16+8h+7
    for (int t = 0; t < KTILES; ++t) {
        const int    code = t * 16 + l16;
        const size_t bb   = (size_t)code * DDIM + bd0;
        AV bHi0, bHi1, bLo0, bLo1;
        bHi0.q[0] = *(const uint4*)(whi + bb);      bHi0.q[1] = *(const uint4*)(whi + bb + 16);
        bHi1.q[0] = *(const uint4*)(whi + bb + 32); bHi1.q[1] = *(const uint4*)(whi + bb + 48);
        bLo0.q[0] = *(const uint4*)(wlo + bb);      bLo0.q[1] = *(const uint4*)(wlo + bb + 16);
        bLo1.q[0] = *(const uint4*)(wlo + bb + 32); bLo1.q[1] = *(const uint4*)(wlo + bb + 48);
        const float c2 = wn2[code];
        if (t + 1 < KTILES) {       // pull next tile of the L2-resident codebook forward
            __builtin_prefetch(whi + bb + 16 * DDIM, 0, 1);
            __builtin_prefetch(wlo + bb + 16 * DDIM, 0, 1);
        }

        v8f acc[2];
        #pragma unroll
        for (int g = 0; g < 2; ++g) {
            v8f a = {0.f, 0.f, 0.f, 0.f, 0.f, 0.f, 0.f, 0.f};
            a = __builtin_amdgcn_wmma_f32_16x16x32_bf16(false, aHi0[g].v, false, bHi0.v, (short)0, a, false, false);
            a = __builtin_amdgcn_wmma_f32_16x16x32_bf16(false, aHi1[g].v, false, bHi1.v, (short)0, a, false, false);
            a = __builtin_amdgcn_wmma_f32_16x16x32_bf16(false, aLo0[g].v, false, bHi0.v, (short)0, a, false, false);
            a = __builtin_amdgcn_wmma_f32_16x16x32_bf16(false, aLo1[g].v, false, bHi1.v, (short)0, a, false, false);
            a = __builtin_amdgcn_wmma_f32_16x16x32_bf16(false, aHi0[g].v, false, bLo0.v, (short)0, a, false, false);
            a = __builtin_amdgcn_wmma_f32_16x16x32_bf16(false, aHi1[g].v, false, bLo1.v, (short)0, a, false, false);
            acc[g] = a;
        }

        #pragma unroll
        for (int g = 0; g < 2; ++g)
            #pragma unroll
            for (int j = 0; j < 8; ++j) {
                const float d2 = rx2[g * 8 + j] + c2 - 2.0f * acc[g][j];  // ||xn||^2+||wn||^2-2x.w
                if (d2 < bestD[g * 8 + j]) { bestD[g * 8 + j] = d2; bestI[g * 8 + j] = code; }
            }
    }

    // ---- argmin across the 16 lanes sharing a row (xor masks stay within half) ----
    #pragma unroll
    for (int j = 0; j < 16; ++j) {
        float d = bestD[j]; int ii = bestI[j];
        #pragma unroll
        for (int m = 1; m < 16; m <<= 1) {
            const float od = __shfl_xor(d, m, 32);
            const int   oi = __shfl_xor(ii, m, 32);
            if (od < d || (od == d && oi < ii)) { d = od; ii = oi; }
        }
        bestD[j] = d; bestI[j] = ii;
    }

    // owners: lanes 0-7 -> in-group rows 0..7, lanes 16-23 -> in-group rows 8..15 (per M-group)
    const bool owner = (lane < 8) || (lane >= 16 && lane < 24);
    float myD[2] = {0.f, 0.f}; int myI[2] = {0, 0};
    #pragma unroll
    for (int g = 0; g < 2; ++g)
        #pragma unroll
        for (int j = 0; j < 8; ++j)
            if ((lane & 7) == j) { myD[g] = bestD[g * 8 + j]; myI[g] = bestI[g * 8 + j]; }
    if (!owner) { myD[0] = myD[1] = 0.f; myI[0] = myI[1] = 0; }
    if (owner) {
        outIdx[rowBlock + wr +      (lane & 7) + lhalf * 8] = (float)myI[0];
        outIdx[rowBlock + wr + 16 + (lane & 7) + lhalf * 8] = (float)myI[1];
    }

    // loss partial: sum of winning d2 ( == ||wn[idx]-xn||^2 ), deterministic order
    float ls = owner ? (myD[0] + myD[1]) : 0.f;
    ls += __shfl_xor(ls, 1, 32); ls += __shfl_xor(ls, 2, 32);
    ls += __shfl_xor(ls, 4, 32); ls += __shfl_xor(ls, 8, 32); ls += __shfl_xor(ls, 16, 32);
    if (lane == 0) lds_ws[wave] = ls;

    // ---- quantized_st = xn + (wn[idx] - xn): gather codebook row, coalesced writes ----
    #pragma unroll
    for (int g = 0; g < 2; ++g) {
        #pragma unroll 4
        for (int r = 0; r < 16; ++r) {
            const int idx = __shfl(myI[g], (r < 8) ? r : r + 8, 32);
            const int rl  = wr + g * 16 + r;
            const int rg  = rowBlock + rl;
            const float xa = lds_xn[rl * DDIM + lane];
            const float xb = lds_xn[rl * DDIM + 32 + lane];
            const float qa = wn[(size_t)idx * DDIM + lane];
            const float qb = wn[(size_t)idx * DDIM + 32 + lane];
            outQ[(size_t)rg * DDIM + lane]      = xa + (qa - xa);
            outQ[(size_t)rg * DDIM + 32 + lane] = xb + (qb - xb);
        }
    }

    __syncthreads();
    if (tid == 0) {
        float t = 0.f;
        for (int wv = 0; wv < WAVES_PER_BLK; ++wv) t += lds_ws[wv];
        partial[blockIdx.x] = t;
    }
}

// ------------- Phase 2: deterministic loss reduction -------------
__global__ void vq_loss(const float* __restrict__ partial, float* __restrict__ outLoss) {
    __shared__ float sm[256];
    const int tid = threadIdx.x;
    float s = 0.f;
    for (int i = tid; i < NBLOCKS; i += 256) s += partial[i];
    sm[tid] = s;
    __syncthreads();
    for (int st = 128; st > 0; st >>= 1) {
        if (tid < st) sm[tid] += sm[tid + st];
        __syncthreads();
    }
    if (tid == 0) {
        const float q = sm[0] / (float)((size_t)N_ROWS * DDIM);  // q_latent == e_latent
        outLoss[0] = q + 0.25f * q;                              // loss = q + beta*e
    }
}

extern "C" void kernel_launch(void* const* d_in, const int* in_sizes, int n_in,
                              void* d_out, int out_size, void* d_ws, size_t ws_size,
                              hipStream_t stream) {
    (void)in_sizes; (void)n_in; (void)out_size; (void)ws_size;
    const float* x = (const float*)d_in[0];        // [N, 64] f32
    const float* w = (const float*)d_in[1];        // [K, 64] f32

    float* out     = (float*)d_out;
    float* outQ    = out;                          // N*64 quantized_st
    float* outLoss = out + (size_t)N_ROWS * DDIM;  // 1
    float* outIdx  = outLoss + 1;                  // N indices (as float)

    // workspace layout (needs ~525 KB)
    float*          wn      = (float*)d_ws;                          // K*64 f32
    float*          wn2     = wn + (size_t)KCODES * DDIM;            // K f32
    unsigned short* whi     = (unsigned short*)(wn2 + KCODES);       // K*64 bf16
    unsigned short* wlo     = whi + (size_t)KCODES * DDIM;           // K*64 bf16
    float*          partial = (float*)(wlo + (size_t)KCODES * DDIM); // NBLOCKS f32

    vq_prep<<<KCODES, 32, 0, stream>>>(w, wn, wn2, whi, wlo);
    vq_main<<<NBLOCKS, 128, 0, stream>>>(x, wn, wn2, whi, wlo, outQ, outIdx, partial);
    vq_loss<<<1, 256, 0, stream>>>(partial, outLoss);
}
